// Head_5153960755560
// MI455X (gfx1250) — matrix operand order repaired and checked by reading, not compiled, not run on hardware
//
#include <hip/hip_runtime.h>
#include <math.h>
#include <stdint.h>

typedef __attribute__((ext_vector_type(16))) __bf16 v16bf;
typedef __attribute__((ext_vector_type(8)))  __bf16 v8bf;
typedef __attribute__((ext_vector_type(8)))  float  v8f;
typedef __attribute__((ext_vector_type(4)))  float  v4f;

#define NB   4          // batch
#define TSEQ 4096       // sequence length
#define CIN  1024       // channels
#define HSD  128        // head size

#define WT_ELEMS  (HSD * CIN)        // one transposed weight, bf16 elems
#define QKV_ELEMS (NB * TSEQ * HSD)  // one projected tensor, bf16 elems

// ---------------------------------------------------------------------------
// helpers
// ---------------------------------------------------------------------------
static __device__ inline v16bf ld_bf16_2x8(const __bf16* p0, const __bf16* p1) {
    // two 16-byte loads -> one 16-element bf16 fragment
    v8bf lo = *(const v8bf*)p0;
    v8bf hi = *(const v8bf*)p1;
    v16bf r;
#pragma unroll
    for (int i = 0; i < 8; ++i) { r[i] = lo[i]; r[i + 8] = hi[i]; }
    return r;
}

static __device__ inline v16bf cvt_f32_2x8(const float* p0, const float* p1) {
    // A-fragment from fp32 source: two contiguous 8-float runs -> bf16
    v4f a0 = *(const v4f*)(p0);
    v4f a1 = *(const v4f*)(p0 + 4);
    v4f b0 = *(const v4f*)(p1);
    v4f b1 = *(const v4f*)(p1 + 4);
    v16bf r;
#pragma unroll
    for (int i = 0; i < 4; ++i) {
        r[i]      = (__bf16)a0[i];
        r[i + 4]  = (__bf16)a1[i];
        r[i + 8]  = (__bf16)b0[i];
        r[i + 12] = (__bf16)b1[i];
    }
    return r;
}

static __device__ inline v8f wmma_bf16(v16bf a, v16bf b, v8f c) {
    return __builtin_amdgcn_wmma_f32_16x16x32_bf16(
        /*neg_a=*/false, a, /*neg_b=*/false, b,
        /*c_mod=*/(short)0, c, /*reuse_a=*/false, /*reuse_b=*/false);
}

// async DMA: 16 bytes global -> LDS, tracked by ASYNCcnt (gfx1250)
static __device__ inline void async_ld16(const __bf16* g, uint32_t lds_off) {
    asm volatile("global_load_async_to_lds_b128 %0, %1, off"
                 :: "v"(lds_off), "v"(g)
                 : "memory");
}
static __device__ inline void wait_asynccnt0() {
    asm volatile("s_wait_asynccnt 0x0" ::: "memory");
}
static __device__ inline uint32_t lds_off_of(const void* p) {
    // flat LDS addresses carry the LDS byte offset in addr[31:0]
    return (uint32_t)(uintptr_t)p;
}

// ---------------------------------------------------------------------------
// Kernel 1: weights fp32 [C,HS] -> bf16 transposed [HS,C], all three at once
// ---------------------------------------------------------------------------
__global__ void wcvt_kernel(const float* __restrict__ Wq,
                            const float* __restrict__ Wk,
                            const float* __restrict__ Wv,
                            __bf16* __restrict__ wt) {
    int idx = blockIdx.x * blockDim.x + threadIdx.x;
    if (idx >= 3 * WT_ELEMS) return;
    int w   = idx / WT_ELEMS;
    int rem = idx - w * WT_ELEMS;
    int n   = rem / CIN;          // output row  (head dim)
    int c   = rem - n * CIN;      // output col  (channel)
    const float* W = (w == 0) ? Wq : (w == 1) ? Wk : Wv;
    wt[idx] = (__bf16)W[c * HSD + n];
}

// ---------------------------------------------------------------------------
// Kernel 2: projection GEMM  [B*T, C] x [C, HS] -> bf16
//   blockIdx.y: 0->q, 1->k, 2->v(transposed)
//   256 threads = 8 waves; wave owns 16 rows x 128 cols.
//   Wt K-chunks (128 x 32, 8 KB) are async-DMA'd into LDS, double-buffered,
//   shared by all 8 waves; x A-fragments stream from global (unique per wave).
// ---------------------------------------------------------------------------

// one Wt chunk: rows 0..127 (head dim) x cols [kc, kc+32)  -> LDS [128][32]
static __device__ inline void issue_wchunk(const __bf16* __restrict__ wt,
                                           int kc, uint32_t w_lds, int tid) {
#pragma unroll
    for (int h = 0; h < 2; ++h) {
        int c = tid + h * 256;                 // chunk id 0..511 (16B chunks)
        const __bf16* g = wt + (size_t)(c >> 2) * CIN + kc + (c & 3) * 8;
        async_ld16(g, w_lds + c * 16);
    }
}

__global__ void proj_kernel(const float*  __restrict__ x,
                            const __bf16* __restrict__ wt3,
                            __bf16* __restrict__ q,
                            __bf16* __restrict__ kk,
                            __bf16* __restrict__ vt) {
    const int which = blockIdx.y;
    const __bf16* wt = wt3 + which * WT_ELEMS;

    const int tid  = threadIdx.x;
    const int wave = tid >> 5;
    const int lane = tid & 31;
    const int m    = lane & 15;
    const int half = lane >> 4;          // 0 or 1
    const int kbA  = half ? 8 : 0;       // A-fragment K base
    const int kbB  = half ? 16 : 0;      // cancelled in B loads; B K base is kbA

    const int rowbase = blockIdx.x * 128 + wave * 16;   // global row in [0, B*T)

    __shared__ __attribute__((aligned(16))) __bf16 wtile[2][128 * 32];  // [dim][k]
    const uint32_t w_lds[2] = { lds_off_of(&wtile[0][0]), lds_off_of(&wtile[1][0]) };

    v8f acc[8];
#pragma unroll
    for (int t = 0; t < 8; ++t)
#pragma unroll
        for (int r = 0; r < 8; ++r) acc[t][r] = 0.0f;

    const float* xrow = x + (size_t)(rowbase + m) * CIN;

    issue_wchunk(wt, 0, w_lds[0], tid);

    for (int it = 0; it < CIN / 32; ++it) {
        const int kc  = it * 32;
        const int buf = it & 1;

        wait_asynccnt0();
        __syncthreads();                           // Wt chunk landed for all

        if (it + 1 < CIN / 32)
            issue_wchunk(wt, kc + 32, w_lds[buf ^ 1], tid);

        v16bf a = cvt_f32_2x8(xrow + kc + kbA, xrow + kc + 16 + kbA);
        const __bf16* wbuf = &wtile[buf][0];
#pragma unroll
        for (int t = 0; t < 8; ++t) {
            const __bf16* bp = wbuf + (size_t)(t * 16 + m) * 32 + kbB;
            v16bf b = ld_bf16_2x8(bp - kbB + kbA, bp - kbB + 16 + kbA);
            acc[t] = wmma_bf16(a, b, acc[t]);
        }

        __syncthreads();                           // reads done before refill
    }

    // store C fragments
#pragma unroll
    for (int t = 0; t < 8; ++t) {
#pragma unroll
        for (int r = 0; r < 8; ++r) {
            int row = rowbase + r + 8 * half;       // global row in [0, B*T)
            int col = t * 16 + m;                   // head dim
            __bf16 val = (__bf16)acc[t][r];
            if (which == 0) {
                q[(size_t)row * HSD + col] = val;
            } else if (which == 1) {
                kk[(size_t)row * HSD + col] = val;
            } else {
                int batch = row >> 12;              // /4096
                int tt    = row & (TSEQ - 1);
                vt[(size_t)batch * HSD * TSEQ + (size_t)col * TSEQ + tt] = val;
            }
        }
    }
}

// ---------------------------------------------------------------------------
// Kernel 3: causal flash attention with async-DMA double-buffered K/V in LDS
//   grid (T/128, B); 256 threads = 8 waves; wave owns 16 query rows
//   block iterates uniformly to its causal limit so barriers stay convergent
// ---------------------------------------------------------------------------

// issue one 32-key tile: K rows [kb,kb+32) x 128 dims  (8 KB, row-major [32][128])
//                        Vt rows 128 dims x keys [kb,kb+32) (8 KB, row-major [128][32])
static __device__ inline void issue_tile(const __bf16* __restrict__ kg,
                                         const __bf16* __restrict__ vg,
                                         int kb, uint32_t k_lds, uint32_t v_lds,
                                         int tid) {
#pragma unroll
    for (int h = 0; h < 2; ++h) {
        int c = tid + h * 256;                     // chunk id 0..511 (16B chunks)
        const __bf16* gk = kg + (size_t)(kb + (c >> 4)) * HSD + (c & 15) * 8;
        async_ld16(gk, k_lds + c * 16);
        const __bf16* gv = vg + (size_t)(c >> 2) * TSEQ + kb + (c & 3) * 8;
        async_ld16(gv, v_lds + c * 16);
    }
}

__global__ void attn_kernel(const __bf16* __restrict__ q,
                            const __bf16* __restrict__ k,
                            const __bf16* __restrict__ vt,
                            float* __restrict__ out) {
    const int tid  = threadIdx.x;
    const int wave = tid >> 5;
    const int lane = tid & 31;
    const int m    = lane & 15;
    const int half = lane >> 4;
    const int kbA  = half ? 8 : 0;
    const int kbB  = half ? 16 : 0;

    const int batch = blockIdx.y;
    const int qbase = blockIdx.x * 128 + wave * 16;
    const int qlast = qbase + 15;

    const __bf16* qb  = q  + (size_t)batch * TSEQ * HSD;
    const __bf16* kg  = k  + (size_t)batch * TSEQ * HSD;
    const __bf16* vg  = vt + (size_t)batch * HSD * TSEQ;

    __shared__ __attribute__((aligned(16))) __bf16 ktile[2][32 * 128];  // [key][dim]
    __shared__ __attribute__((aligned(16))) __bf16 vtile[2][128 * 32];  // [dim][key]
    __shared__ __attribute__((aligned(16))) __bf16 pstage[8][16][32];

    const uint32_t k_lds[2] = { lds_off_of(&ktile[0][0]), lds_off_of(&ktile[1][0]) };
    const uint32_t v_lds[2] = { lds_off_of(&vtile[0][0]), lds_off_of(&vtile[1][0]) };

    // Q fragments: 16 rows x 128 dims = 4 A-fragments (K chunks of 32)
    v16bf af[4];
#pragma unroll
    for (int c = 0; c < 4; ++c) {
        const __bf16* qp = qb + (size_t)(qbase + m) * HSD + c * 32;
        af[c] = ld_bf16_2x8(qp + kbA, qp + 16 + kbA);
    }

    v8f o[8];
    float mst[8], lst[8];
#pragma unroll
    for (int t = 0; t < 8; ++t)
#pragma unroll
        for (int r = 0; r < 8; ++r) o[t][r] = 0.0f;
#pragma unroll
    for (int r = 0; r < 8; ++r) { mst[r] = -1e30f; lst[r] = 0.0f; }

    const float scale = 0.08838834764831845f;   // 128^-0.5
    const int nit = blockIdx.x * 4 + 4;         // key blocks up to block causal limit

    issue_tile(kg, vg, 0, k_lds[0], v_lds[0], tid);

    for (int it = 0; it < nit; ++it) {
        const int kb  = it * 32;
        const int buf = it & 1;

        wait_asynccnt0();          // this thread's DMA for buf has landed
        __syncthreads();           // ... and everyone else's

        if (it + 1 < nit)          // prefetch next tile into the other buffer
            issue_tile(kg, vg, kb + 32, k_lds[buf ^ 1], v_lds[buf ^ 1], tid);

        if (kb <= qlast) {
            const __bf16* kbuf = &ktile[buf][0];
            const __bf16* vbuf = &vtile[buf][0];

            // ---- S = Q K^T for keys [kb, kb+32) : two 16x16 C fragments ----
            v8f s0, s1;
#pragma unroll
            for (int r = 0; r < 8; ++r) { s0[r] = 0.0f; s1[r] = 0.0f; }
#pragma unroll
            for (int c = 0; c < 4; ++c) {
                const __bf16* kp0 = kbuf + (size_t)m * HSD        + c * 32 + kbB;
                const __bf16* kp1 = kbuf + (size_t)(16 + m) * HSD + c * 32 + kbB;
                v16bf b0 = ld_bf16_2x8(kp0 - kbB + kbA, kp0 - kbB + 16 + kbA);
                v16bf b1 = ld_bf16_2x8(kp1 - kbB + kbA, kp1 - kbB + 16 + kbA);
                s0 = wmma_bf16(af[c], b0, s0);
                s1 = wmma_bf16(af[c], b1, s1);
            }
#pragma unroll
            for (int r = 0; r < 8; ++r) { s0[r] *= scale; s1[r] *= scale; }

            // ---- causal mask (only needed near the diagonal) ----
            if (kb + 31 > qbase) {
#pragma unroll
                for (int r = 0; r < 8; ++r) {
                    int rowg = qbase + r + 8 * half;
                    if (kb + m      > rowg) s0[r] = -1e30f;
                    if (kb + 16 + m > rowg) s1[r] = -1e30f;
                }
            }

            // ---- online softmax (each row lives in one 16-lane half) ----
            float pr0[8], pr1[8], alpha[8];
#pragma unroll
            for (int r = 0; r < 8; ++r) {
                float mx = fmaxf(s0[r], s1[r]);
#pragma unroll
                for (int d = 1; d < 16; d <<= 1) mx = fmaxf(mx, __shfl_xor(mx, d, 32));
                float mnew = fmaxf(mst[r], mx);
                alpha[r] = __expf(mst[r] - mnew);
                mst[r] = mnew;
                pr0[r] = __expf(s0[r] - mnew);
                pr1[r] = __expf(s1[r] - mnew);
                float rs = pr0[r] + pr1[r];
#pragma unroll
                for (int d = 1; d < 16; d <<= 1) rs += __shfl_xor(rs, d, 32);
                lst[r] = lst[r] * alpha[r] + rs;
            }
#pragma unroll
            for (int t = 0; t < 8; ++t)
#pragma unroll
                for (int r = 0; r < 8; ++r) o[t][r] *= alpha[r];

            // ---- stage P (C layout) -> LDS -> A-fragment layout ----
            __builtin_amdgcn_wave_barrier();
#pragma unroll
            for (int r = 0; r < 8; ++r) {
                int row = r + 8 * half;
                pstage[wave][row][m]      = (__bf16)pr0[r];
                pstage[wave][row][16 + m] = (__bf16)pr1[r];
            }
            __builtin_amdgcn_wave_barrier();
            v16bf pa = ld_bf16_2x8(&pstage[wave][m][kbA], &pstage[wave][m][16 + kbA]);
            __builtin_amdgcn_wave_barrier();

            // ---- O += P * V   (Vt tile: contiguous along keys) ----
#pragma unroll
            for (int t = 0; t < 8; ++t) {
                const __bf16* vp = vbuf + (size_t)(t * 16 + m) * 32 + kbB;
                v16bf bv = ld_bf16_2x8(vp - kbB + kbA, vp - kbB + 16 + kbA);
                o[t] = wmma_bf16(pa, bv, o[t]);
            }
        }

        __syncthreads();           // reads of buf done before it is refilled
    }

    // ---- normalize and store fp32 output ----
#pragma unroll
    for (int t = 0; t < 8; ++t) {
#pragma unroll
        for (int r = 0; r < 8; ++r) {
            int row = qbase + r + 8 * half;
            int col = t * 16 + m;
            out[((size_t)batch * TSEQ + row) * HSD + col] = o[t][r] / lst[r];
        }
    }
}

// ---------------------------------------------------------------------------
// launcher
// ---------------------------------------------------------------------------
extern "C" void kernel_launch(void* const* d_in, const int* in_sizes, int n_in,
                              void* d_out, int out_size, void* d_ws, size_t ws_size,
                              hipStream_t stream) {
    const float* x  = (const float*)d_in[0];
    const float* Wq = (const float*)d_in[1];
    const float* Wk = (const float*)d_in[2];
    const float* Wv = (const float*)d_in[3];
    float* out = (float*)d_out;

    __bf16* ws  = (__bf16*)d_ws;
    __bf16* wt3 = ws;                          // 3 * 128K bf16
    __bf16* qb  = ws + 3 * WT_ELEMS;           // 2M bf16
    __bf16* kb  = qb + QKV_ELEMS;              // 2M bf16
    __bf16* vtb = kb + QKV_ELEMS;              // 2M bf16

    // 1) weight transpose + bf16 convert
    {
        int total = 3 * WT_ELEMS;
        wcvt_kernel<<<(total + 255) / 256, 256, 0, stream>>>(Wq, Wk, Wv, wt3);
    }
    // 2) q/k/v projections (Wt chunks async-DMA'd to LDS; v stored transposed)
    proj_kernel<<<dim3((NB * TSEQ) / 128, 3), 256, 0, stream>>>(x, wt3, qb, kb, vtb);
    // 3) causal flash attention (async-DMA double-buffered K/V)
    attn_kernel<<<dim3(TSEQ / 128, NB), 256, 0, stream>>>(qb, kb, vtb, out);
}
